// Generator_12876311953953
// MI455X (gfx1250) — matrix operand order repaired and checked by reading, not compiled, not run on hardware
//
#include <hip/hip_runtime.h>
#include <hip/hip_bf16.h>

// ---------------------------------------------------------------------------
// StyleGAN2 generator for MI455X (gfx1250, wave32, WMMA).
// Heavy convs as implicit GEMM on v_wmma_f32_16x16x32_f16:
//   - 64-pixel N tile per block (4 WMMA subtiles per wave) so each packed
//     A fragment is reused 4x -> A traffic ~0.6 GB, L2-resident
//   - packed A fragments double-buffered through LDS via
//     global_load_async_to_lds_b128 + s_wait_asynccnt (ASYNCcnt)
//   - modulated f16 im2col B tiles staged in LDS (conflict-free stride)
// Demod GEMM on v_wmma_f32_16x16x4_f32.
// ---------------------------------------------------------------------------

#define MODCONV_ASYNC 1

typedef __attribute__((ext_vector_type(16))) _Float16 v16h;
typedef __attribute__((ext_vector_type(8)))  float    v8f;
typedef __attribute__((ext_vector_type(2)))  float    v2f;

// ===========================================================================
// GEMM16: C(16 x N) = act( A(16 x K) @ Wt(N,K)^T + bias )
// One wave per 16-col tile. mode 0: lrelu (mapping MLP); mode 1: +1 (styles).
// ===========================================================================
__global__ __launch_bounds__(32)
void gemm16_kernel(const float* __restrict__ A, const float* __restrict__ Wt,
                   const float* __restrict__ bias, float* __restrict__ out,
                   int K, int N, int mode) {
    const int lane = threadIdx.x;
    const int n0   = blockIdx.x * 16;
    const int m    = lane & 15;          // A row (batch)
    const int half = lane >> 4;
    const int n    = n0 + (lane & 15);   // B col
    const int kbA  = half * 8;
    const int kbB  = half * 16;

    v8f c = {};
    const int nkc = K >> 5;
    for (int kc = 0; kc < nkc; ++kc) {
        v16h a, b;
        const int k0 = kc * 32;
        #pragma unroll
        for (int j = 0; j < 16; ++j) {
            const int ka = k0 + half * 16 + j;
            a[j] = (_Float16)A[m * K + ka];
        }
        const float* wp = Wt + (size_t)n * K + k0 + kbB;
        #pragma unroll
        for (int j = 0; j < 16; ++j) b[j] = (_Float16)wp[j];
        c = __builtin_amdgcn_wmma_f32_16x16x32_f16(false, a, false, b,
                                                   (short)0, c, false, false);
    }
    #pragma unroll
    for (int r = 0; r < 8; ++r) {
        const int mm = r + half * 8;
        float v = c[r] + bias[n];
        if (mode == 0) v = v > 0.f ? v : 0.2f * v;   // leaky relu
        else           v += 1.0f;                    // style: +1
        out[mm * N + n] = v;
    }
}

// ===========================================================================
// pixel norm: w = h * rsqrt(mean(h^2) + 1e-8), h is (16,512)
// ===========================================================================
__global__ __launch_bounds__(256)
void pixelnorm_kernel(const float* __restrict__ h, float* __restrict__ w) {
    __shared__ float red[256];
    const int b = blockIdx.x, t = threadIdx.x;
    const float v0 = h[b * 512 + t];
    const float v1 = h[b * 512 + 256 + t];
    red[t] = v0 * v0 + v1 * v1;
    __syncthreads();
    for (int s = 128; s > 0; s >>= 1) {
        if (t < s) red[t] += red[t + s];
        __syncthreads();
    }
    const float sc = rsqrtf(red[0] * (1.0f / 512.0f) + 1e-8f);
    w[b * 512 + t]       = v0 * sc;
    w[b * 512 + 256 + t] = v1 * sc;
}

// ===========================================================================
// Pack conv weight (co,ci,3,3) f32 -> f16 in WMMA A-fragment order.
// ===========================================================================
__global__ __launch_bounds__(32)
void pack_kernel(const float* __restrict__ Wsrc, _Float16* __restrict__ packed,
                 int ci, int nkc) {
    const int lane = threadIdx.x;
    const int ot = blockIdx.x, kc = blockIdx.y;
    const int m  = ot * 16 + (lane & 15);
    const int kb = (lane >> 4) * 8;
    v16h a;
    #pragma unroll
    for (int j = 0; j < 16; ++j) {
        const int k = kc * 32 + (lane >> 4) * 16 + j;
        const int i = k / 9;
        const int t = k - i * 9;
        a[j] = (_Float16)Wsrc[((size_t)m * ci + i) * 9 + t];
    }
    *reinterpret_cast<v16h*>(packed + (((size_t)ot * nkc + kc) * 32 + lane) * 16) = a;
}

// ===========================================================================
// w2[o,i] = sum_{k,l} W[o,i,k,l]^2
// ===========================================================================
__global__ __launch_bounds__(256)
void w2_kernel(const float* __restrict__ Wsrc, float* __restrict__ w2, int total) {
    const int idx = blockIdx.x * 256 + threadIdx.x;
    if (idx >= total) return;
    const float* p = Wsrc + (size_t)idx * 9;
    float s = 0.f;
    #pragma unroll
    for (int j = 0; j < 9; ++j) s += p[j] * p[j];
    w2[idx] = s;
}

// ===========================================================================
// demod: d[b,o] = rsqrt( sum_i s[b,i]^2 * w2[o,i] + 1e-8 )
// GEMM M=16(batch) x N=co x K=ci on v_wmma_f32_16x16x4_f32.
// ===========================================================================
__global__ __launch_bounds__(32)
void demod_kernel(const float* __restrict__ smod, const float* __restrict__ w2,
                  float* __restrict__ dmod, int ci, int co) {
    const int lane = threadIdx.x;
    const int n0   = blockIdx.x * 16;
    const int m    = lane & 15;
    const int half = lane >> 4;
    const int n    = n0 + (lane & 15);
#if __has_builtin(__builtin_amdgcn_wmma_f32_16x16x4_f32)
    v8f c = {};
    const int nkc = ci >> 2;
    for (int kc = 0; kc < nkc; ++kc) {
        v2f a, b;
        #pragma unroll
        for (int r = 0; r < 2; ++r) {
            const int k = kc * 4 + half * 2 + r;
            const float sv = smod[m * ci + k];
            a[r] = sv * sv;
            b[r] = w2[(size_t)n * ci + k];
        }
        c = __builtin_amdgcn_wmma_f32_16x16x4_f32(false, a, false, b,
                                                  (short)0, c, false, false);
    }
    #pragma unroll
    for (int r = 0; r < 8; ++r) {
        const int mm = r + half * 8;
        dmod[mm * co + n] = rsqrtf(c[r] + 1e-8f);
    }
#else
    for (int r = 0; r < 8; ++r) {
        const int mm = r + half * 8;
        float acc = 0.f;
        for (int k = 0; k < ci; ++k) {
            const float sv = smod[mm * ci + k];
            acc += sv * sv * w2[(size_t)n * ci + k];
        }
        dmod[mm * co + n] = rsqrtf(acc + 1e-8f);
    }
#endif
}

// ===========================================================================
// Modulated 3x3 conv (implicit GEMM), optional fused nearest 2x upsample.
// 128 threads = 4 waves; block owns (batch b, 64-pixel group); wave w owns
// output-channel tile ot = blockIdx.y*4 + w and computes FOUR 16x16 WMMA
// subtiles from one shared A fragment per K-chunk (4x A reuse).
// Per K-chunk of 32:
//   - all threads build the modulated f16 im2col tile (32K x 64px) in LDS
//   - each wave's packed A fragment (1KB) is double-buffered through LDS
//     with global_load_async_to_lds_b128 issued one chunk ahead, drained
//     with s_wait_asynccnt before the ds_load_b128 reads
//   - 4x v_wmma_f32_16x16x32_f16 accumulate the 16x64 f32 tile
// Epilogue: *demod, + nscale*noise, leaky relu.
// ===========================================================================
__global__ __launch_bounds__(128)
void modconv_kernel(const float* __restrict__ src, unsigned long long srcBStride,
                    const _Float16* __restrict__ packedA,
                    const float* __restrict__ smod,
                    const float* __restrict__ dmod,
                    const float* __restrict__ nscale,
                    const float* __restrict__ noise,
                    float* __restrict__ out,
                    int ci, int co, int W, int logW, int up, int nkc) {
    __shared__ __attribute__((aligned(16))) _Float16 Bld[64 * 40];   // 5 KB
#if MODCONV_ASYNC
    __shared__ __attribute__((aligned(16))) _Float16 Ald[4][2][512]; // 8 KB
#endif

    const int tid    = threadIdx.x;
    const int lane   = tid & 31;
    const int wave   = tid >> 5;
    const int pgrps  = (W * W) >> 6;             // 64-pixel groups
    const int b      = blockIdx.x / pgrps;
    const int pg     = blockIdx.x - b * pgrps;
    const int ot     = blockIdx.y * 4 + wave;

    const int srcW = up ? (W >> 1) : W;
    const float* srcb = src + (size_t)b * srcBStride;
    const float* sb   = smod + b * ci;

    const int half = lane >> 4;
    const int nB   = lane & 15;
    const int kbB  = half * 16;

#if MODCONV_ASYNC
    // per-lane global source of this wave's A fragment stream (bytes)
    const unsigned long long gfrag =
        (unsigned long long)(uintptr_t)packedA +
        (unsigned long long)ot * nkc * 1024ULL + (unsigned long long)lane * 16ULL;
    // per-lane LDS destinations (byte-image copy: lane covers [lane*16,+16)
    // of each 512B half; offset:512 covers the second half)
    const unsigned dstLds[2] = {
        (unsigned)(uintptr_t)(&Ald[wave][0][0]) + (unsigned)lane * 16u,
        (unsigned)(uintptr_t)(&Ald[wave][1][0]) + (unsigned)lane * 16u };

    // prologue: async-stage fragment kc=0 into buffer 0
    asm volatile("global_load_async_to_lds_b128 %0, %1, off\n\t"
                 "global_load_async_to_lds_b128 %0, %1, off offset:512"
                 :: "v"(dstLds[0]), "v"(gfrag) : "memory");
#else
    const _Float16* aptr = packedA + (((size_t)ot * nkc) * 32 + lane) * 16;
#endif

    v8f c0 = {}, c1 = {}, c2 = {}, c3 = {};
    for (int kc = 0; kc < nkc; ++kc) {
#if MODCONV_ASYNC
        // issue next fragment into the other buffer (clamped on last iter;
        // targets the buffer we are NOT about to read)
        {
            const int knext = (kc + 1 < nkc) ? (kc + 1) : kc;
            const unsigned long long ga = gfrag + (unsigned long long)knext * 1024ULL;
            asm volatile("global_load_async_to_lds_b128 %0, %1, off\n\t"
                         "global_load_async_to_lds_b128 %0, %1, off offset:512"
                         :: "v"(dstLds[(kc + 1) & 1]), "v"(ga) : "memory");
        }
#endif
        // ---- stage modulated im2col chunk (32 K x 64 pixels) into LDS ----
        #pragma unroll
        for (int r = 0; r < 16; ++r) {
            const int e  = tid + 128 * r;
            const int kk = e >> 6;               // 0..31
            const int n  = e & 63;               // 0..63
            const int k  = kc * 32 + kk;
            const int i  = k / 9;
            const int t  = k - i * 9;
            const int kh = t / 3;
            const int kw = t - kh * 3;
            const int p  = pg * 64 + n;
            const int oh = p >> logW;
            const int ow = p & (W - 1);
            const int ihu = oh + kh - 1;
            const int iwu = ow + kw - 1;
            float v = 0.f;
            if (ihu >= 0 && ihu < W && iwu >= 0 && iwu < W) {
                const int sh = up ? (ihu >> 1) : ihu;
                const int sw = up ? (iwu >> 1) : iwu;
                v = srcb[((size_t)i * srcW + sh) * srcW + sw] * sb[i];
            }
            Bld[n * 40 + kk] = (_Float16)v;
        }
        __syncthreads();

#if MODCONV_ASYNC
        // drain: with <=4 outstanding and in-order completion, asynccnt<=2
        // guarantees the 2 copies of the current chunk have landed in LDS
        asm volatile("s_wait_asynccnt 0x2" ::: "memory");
        union { uint4 q[2]; v16h v; } au;
        const _Float16* ap = &Ald[wave][kc & 1][0] + lane * 16;
        au.q[0] = *reinterpret_cast<const uint4*>(ap);
        au.q[1] = *reinterpret_cast<const uint4*>(ap + 8);
        const v16h a = au.v;
#else
        v16h a = *reinterpret_cast<const v16h*>(aptr + (size_t)kc * 32 * 16);
        __builtin_prefetch(aptr + (size_t)(kc + 1) * 32 * 16, 0, 0);
#endif

        // ---- 4 B subtiles from LDS, 4 WMMAs sharing one A fragment ----
        #pragma unroll
        for (int s = 0; s < 4; ++s) {
            union { uint4 q[2]; v16h v; } bu;
            const _Float16* bp = &Bld[(s * 16 + nB) * 40 + kbB];
            bu.q[0] = *reinterpret_cast<const uint4*>(bp);
            bu.q[1] = *reinterpret_cast<const uint4*>(bp + 8);
            v8f* acc = (s == 0) ? &c0 : (s == 1) ? &c1 : (s == 2) ? &c2 : &c3;
            *acc = __builtin_amdgcn_wmma_f32_16x16x32_f16(false, a, false, bu.v,
                                                          (short)0, *acc, false, false);
        }
        __syncthreads();
    }

    // ---- epilogue: demod, noise, leaky relu, store NCHW f32 ----
    #pragma unroll
    for (int s = 0; s < 4; ++s) {
        const v8f& c = (s == 0) ? c0 : (s == 1) ? c1 : (s == 2) ? c2 : c3;
        const int p  = pg * 64 + s * 16 + nB;
        const int oh = p >> logW;
        const int ow = p & (W - 1);
        #pragma unroll
        for (int r = 0; r < 8; ++r) {
            const int m = r + half * 8;
            const int o = ot * 16 + m;
            float v = c[r];
            v *= dmod[b * co + o];
            const size_t oidx = (((size_t)b * co + o) * W + oh) * W + ow;
            v += nscale[o] * noise[oidx];
            v = v > 0.f ? v : 0.2f * v;
            out[oidx] = v;
        }
    }
}

// ===========================================================================
// toRGB 1x1 modulated conv (no demod), optional += up2(prev), optional tanh.
// ===========================================================================
__global__ __launch_bounds__(256)
void rgb_kernel(const float* __restrict__ x, const float* __restrict__ smod,
                const float* __restrict__ tw, const float* __restrict__ prev,
                float* __restrict__ out, int ci, int W, int dotanh) {
    const int idx = blockIdx.x * 256 + threadIdx.x;
    const int P = W * W;
    const int total = 16 * 3 * P;
    if (idx >= total) return;
    const int p = idx % P;
    const int c = (idx / P) % 3;
    const int b = idx / (3 * P);
    const float* xb  = x + (size_t)b * ci * P + p;
    const float* sbb = smod + b * ci;
    const float* twc = tw + c * ci;
    float acc = 0.f;
    for (int i = 0; i < ci; ++i)
        acc += xb[(size_t)i * P] * sbb[i] * twc[i];
    if (prev) {
        const int oh = p / W, ow = p - (p / W) * W;
        const int Wp = W >> 1;
        acc += prev[((size_t)(b * 3 + c) * Wp + (oh >> 1)) * Wp + (ow >> 1)];
    }
    if (dotanh) acc = tanhf(acc);
    out[idx] = acc;
}

// ===========================================================================
// host launch
// ===========================================================================
extern "C" void kernel_launch(void* const* d_in, const int* in_sizes, int n_in,
                              void* d_out, int out_size, void* d_ws, size_t ws_size,
                              hipStream_t stream) {
    (void)in_sizes; (void)n_in; (void)out_size; (void)ws_size;

    const float* z      = (const float*)d_in[0];
    const float* map_w0 = (const float*)d_in[1];
    const float* map_b0 = (const float*)d_in[2];
    const float* map_w  = (const float*)d_in[3];
    const float* map_b  = (const float*)d_in[4];
    const float* cnst   = (const float*)d_in[5];

    // workspace bump allocator (256B aligned)
    char* ws = (char*)d_ws;
    size_t off = 0;
    auto alloc = [&](size_t bytes) -> char* {
        off = (off + 255) & ~(size_t)255;
        char* p = ws + off;
        off += bytes;
        return p;
    };

    float* hA = (float*)alloc(16 * 512 * 4);
    float* hB = (float*)alloc(16 * 512 * 4);
    float* wv = (float*)alloc(16 * 512 * 4);

    const int ciL[6] = {2048, 1024, 1024, 512, 512, 256};
    const int coL[6] = {1024, 1024,  512, 512, 256, 256};
    const int WL[6]  = {   8,    8,   16,  16,  32,  32};
    const int lgW[6] = {   3,    3,    4,   4,   5,   5};
    const int upL[6] = {   1,    0,    1,   0,   1,   0};
    const int wIdx[6]  = {6, 9, 19, 22, 32, 35};
    const int awIdx[6] = {7, 10, 20, 23, 33, 36};
    const int abIdx[6] = {8, 11, 21, 24, 34, 37};
    const int nsIdx[6] = {12, 13, 25, 26, 38, 39};
    const int nzIdx[6] = {14, 15, 27, 28, 40, 41};
    const int tciL[3]  = {1024, 512, 256};
    const int twIdx[3] = {16, 29, 42};
    const int tawIdx[3] = {17, 30, 43};
    const int tabIdx[3] = {18, 31, 44};

    float* sC[6]; float* dC[6];
    for (int k = 0; k < 6; ++k) sC[k] = (float*)alloc((size_t)16 * ciL[k] * 4);
    for (int k = 0; k < 6; ++k) dC[k] = (float*)alloc((size_t)16 * coL[k] * 4);
    float* sT[3];
    for (int t = 0; t < 3; ++t) sT[t] = (float*)alloc((size_t)16 * tciL[t] * 4);

    float*     w2buf  = (float*)alloc((size_t)1024 * 2048 * 4);          // 8 MB
    _Float16*  packed = (_Float16*)alloc((size_t)1024 * 2048 * 9 * 2);   // 38 MB
    float* x0 = (float*)alloc((size_t)16 * 256 * 32 * 32 * 4);           // 16 MB
    float* x1 = (float*)alloc((size_t)16 * 256 * 32 * 32 * 4);           // 16 MB
    float* rgb0 = (float*)alloc((size_t)16 * 3 * 64 * 4);
    float* rgb1 = (float*)alloc((size_t)16 * 3 * 256 * 4);

    // ---- mapping network: 8 lrelu layers ----
    gemm16_kernel<<<32, 32, 0, stream>>>(z, map_w0, map_b0, hA, 128, 512, 0);
    float* cur = hA; float* nxt = hB;
    for (int i = 0; i < 7; ++i) {
        gemm16_kernel<<<32, 32, 0, stream>>>(cur, map_w + (size_t)i * 512 * 512,
                                             map_b + i * 512, nxt, 512, 512, 0);
        float* t = cur; cur = nxt; nxt = t;
    }
    pixelnorm_kernel<<<16, 256, 0, stream>>>(cur, wv);

    // ---- style affines ----
    for (int k = 0; k < 6; ++k)
        gemm16_kernel<<<ciL[k] / 16, 32, 0, stream>>>(
            wv, (const float*)d_in[awIdx[k]], (const float*)d_in[abIdx[k]],
            sC[k], 512, ciL[k], 1);
    for (int t = 0; t < 3; ++t)
        gemm16_kernel<<<tciL[t] / 16, 32, 0, stream>>>(
            wv, (const float*)d_in[tawIdx[t]], (const float*)d_in[tabIdx[t]],
            sT[t], 512, tciL[t], 1);

    // ---- synthesis: 6 modulated convs + 3 toRGB ----
    const float* srcs[6] = {cnst, x0, x1, x0, x1, x0};
    float*       outs[6] = {x0, x1, x0, x1, x0, x1};

    for (int k = 0; k < 6; ++k) {
        const int ci = ciL[k], co = coL[k], W = WL[k], up = upL[k];
        const int nkc = ci * 9 / 32;
        const float* Wk = (const float*)d_in[wIdx[k]];

        pack_kernel<<<dim3(co / 16, nkc), 32, 0, stream>>>(Wk, packed, ci, nkc);
        const int tot = co * ci;
        w2_kernel<<<(tot + 255) / 256, 256, 0, stream>>>(Wk, w2buf, tot);
        demod_kernel<<<co / 16, 32, 0, stream>>>(sC[k], w2buf, dC[k], ci, co);

        const int srcH = up ? (W >> 1) : W;
        const unsigned long long bstr =
            (k == 0) ? 0ULL : (unsigned long long)ci * srcH * srcH;
        const int pgrps = (W * W) / 64;
        modconv_kernel<<<dim3(pgrps * 16, co / 64), 128, 0, stream>>>(
            srcs[k], bstr, packed, sC[k], dC[k],
            (const float*)d_in[nsIdx[k]], (const float*)d_in[nzIdx[k]],
            outs[k], ci, co, W, lgW[k], up, nkc);

        if (k == 1)
            rgb_kernel<<<(16 * 3 * 64 + 255) / 256, 256, 0, stream>>>(
                x1, sT[0], (const float*)d_in[twIdx[0]], nullptr, rgb0, 1024, 8, 0);
        if (k == 3)
            rgb_kernel<<<(16 * 3 * 256 + 255) / 256, 256, 0, stream>>>(
                x1, sT[1], (const float*)d_in[twIdx[1]], rgb0, rgb1, 512, 16, 0);
        if (k == 5)
            rgb_kernel<<<(16 * 3 * 1024 + 255) / 256, 256, 0, stream>>>(
                x1, sT[2], (const float*)d_in[twIdx[2]], rgb1, (float*)d_out, 256, 32, 1);
    }
}
